// ScatterMoE_66271345377807
// MI455X (gfx1250) — compile-verified, matrix operand order checked by reading, not run on hardware
//
#include <hip/hip_runtime.h>

// ---------------------------------------------------------------------------
// MoE (top-2, gated SiLU FFN) for gfx1250 / MI455X.
// Strategy: route -> bucket tokens per expert -> per-expert bf16 WMMA GEMMs.
// Matrix math via v_wmma_f32_16x16x32_bf16 (wave32); phase-2 A tiles staged
// to LDS by the Tensor Data Mover (tensor_load_to_lds + s_wait_tensorcnt).
// ---------------------------------------------------------------------------

typedef __attribute__((ext_vector_type(16))) __bf16 v16bf;
typedef __attribute__((ext_vector_type(8)))  float  v8f;
typedef __attribute__((ext_vector_type(4)))  unsigned int v4u;
typedef __attribute__((ext_vector_type(8)))  int v8i;
typedef __attribute__((ext_vector_type(4)))  int v4i;
typedef unsigned int u32;

union Frag { v16bf v; uint4 q[2]; };

#define T_TOK 4096
#define DDIM  1024
#define IDIM  1024
#define NEXP  8
#define LSTR  48   // LDS row stride in bf16 elements (96B: 16B-aligned, bank-spread)

#if defined(__has_builtin)
#if __has_builtin(__builtin_amdgcn_tensor_load_to_lds) && \
    __has_builtin(__builtin_amdgcn_s_wait_tensorcnt)
#define USE_TDM 1
#endif
#endif
#ifndef USE_TDM
#define USE_TDM 0
#endif

__device__ __forceinline__ unsigned short f2bf(float f) {
  unsigned int u = __float_as_uint(f);
  u += 0x7FFFu + ((u >> 16) & 1u);   // round-to-nearest-even
  return (unsigned short)(u >> 16);
}
__device__ __forceinline__ unsigned int pk2(float a, float b) {
  return (unsigned int)f2bf(a) | ((unsigned int)f2bf(b) << 16);
}

#if USE_TDM
// TDM: copy a 128-row x 64-byte 2D tile (bf16, viewed as 16 DWORDs/row) from
// global to LDS, padding 8 DWORDs (32B) after every 16 DWORDs so LDS rows land
// on the 96B fragment stride. Rows beyond rows_avail are zero-filled by the
// TDM's tensor-dim OOB handling.
__device__ __forceinline__ void tdm_load_tileA(u32 lds_byte_addr,
                                               const void* gptr,
                                               u32 rows_avail) {
  unsigned long long ga = (unsigned long long)(size_t)gptr;
  v4u g0;
  g0[0] = 1u;                                   // count=1, user descriptor
  g0[1] = lds_byte_addr;                        // LDS dest (bytes)
  g0[2] = (u32)(ga & 0xFFFFFFFFu);              // global_addr[31:0]
  g0[3] = (u32)((ga >> 32) & 0x01FFFFFFu) | (2u << 30);  // addr[56:32], type=2

  const u32 dim0 = IDIM / 2;                    // 512 DWORDs per row
  v8i g1;
  g1[0] = (int)((2u << 16)       // data_size = 4B
              | (1u << 20)       // pad_enable
              | (3u << 22)       // pad_interval: 16 DWORDs
              | (7u << 25));     // pad_amount: 8 DWORDs (32B)
  g1[1] = (int)((dim0 & 0xFFFFu) << 16);                    // tensor_dim0 lo16
  g1[2] = (int)(((dim0 >> 16) & 0xFFFFu) |
                ((rows_avail & 0xFFFFu) << 16));            // dim0 hi | dim1 lo
  g1[3] = (int)(((rows_avail >> 16) & 0xFFFFu) |
                (16u << 16));                               // dim1 hi | tile_dim0=16 DW
  g1[4] = (int)128u;                                        // tile_dim1=128 rows
  g1[5] = (int)dim0;                                        // tensor_dim0_stride lo32
  g1[6] = 0;
  g1[7] = 0;
  v4i z4 = {0, 0, 0, 0};
#if defined(__clang_major__) && (__clang_major__ >= 23)
  v8i z8 = {0, 0, 0, 0, 0, 0, 0, 0};
  __builtin_amdgcn_tensor_load_to_lds(g0, g1, z4, z4, z8, 0);
#else
  __builtin_amdgcn_tensor_load_to_lds(g0, g1, z4, z4, 0);
#endif
}
#endif  // USE_TDM

// ---------------------------------------------------------------------------
// Router: one wave32 per token. logits = x . router_w[e], softmax, top-2.
// ---------------------------------------------------------------------------
__global__ void router_kernel(const float* __restrict__ x,
                              const float* __restrict__ rw,
                              float* __restrict__ logits_out,
                              float* __restrict__ wbuf,
                              int* __restrict__ bucket,
                              int* __restrict__ counts) {
  const int lane = threadIdx.x & 31;
  const int wv   = threadIdx.x >> 5;
  const int t    = blockIdx.x * 8 + wv;

  float s[NEXP];
#pragma unroll
  for (int e = 0; e < NEXP; ++e) s[e] = 0.f;

  const float* xp = x + (size_t)t * DDIM;
  for (int d = lane; d < DDIM; d += 32) {
    float xv = xp[d];
#pragma unroll
    for (int e = 0; e < NEXP; ++e) s[e] += xv * rw[e * DDIM + d];
  }
#pragma unroll
  for (int e = 0; e < NEXP; ++e)
    for (int off = 16; off; off >>= 1) s[e] += __shfl_xor(s[e], off);

  if (lane == 0) {
    float m = s[0];
#pragma unroll
    for (int e = 1; e < NEXP; ++e) m = fmaxf(m, s[e]);
    float p[NEXP];
#pragma unroll
    for (int e = 0; e < NEXP; ++e) {
      logits_out[t * NEXP + e] = s[e];
      p[e] = __expf(s[e] - m);
    }
    int e0 = 0; float b0 = p[0];
#pragma unroll
    for (int e = 1; e < NEXP; ++e) if (p[e] > b0) { b0 = p[e]; e0 = e; }
    int e1 = -1; float b1 = -1.f;
#pragma unroll
    for (int e = 0; e < NEXP; ++e) if (e != e0 && p[e] > b1) { b1 = p[e]; e1 = e; }
    float inv = 1.f / (b0 + b1);
    wbuf[t * 2 + 0] = b0 * inv;
    wbuf[t * 2 + 1] = b1 * inv;
    int p0 = atomicAdd(&counts[e0], 1);
    bucket[e0 * T_TOK + p0] = (t << 1);
    int p1 = atomicAdd(&counts[e1], 1);
    bucket[e1 * T_TOK + p1] = (t << 1) | 1;
  }
}

__global__ void init_kernel(int* __restrict__ counts) {
  if (threadIdx.x < NEXP) counts[threadIdx.x] = 0;
}

__global__ void scan_kernel(const int* __restrict__ counts,
                            int* __restrict__ offsets) {
  if (threadIdx.x == 0) {
    int o = 0;
    for (int e = 0; e < NEXP; ++e) { offsets[e] = o; o += counts[e]; }
  }
}

// ---------------------------------------------------------------------------
// Phase 1: for each expert, A = silu(X W1) * (X W3), A stored bf16 compact.
// Block tile 128(M) x 64(N), K-step 32, 8 waves (4x2), wave tile 32x32.
// ---------------------------------------------------------------------------
__device__ __forceinline__ void stage1(unsigned short* sA, unsigned short* sB1,
                                       unsigned short* sB3,
                                       const float* __restrict__ x,
                                       const float* __restrict__ w1p,
                                       const float* __restrict__ w3p,
                                       const int* __restrict__ bucket,
                                       int e, int cnt, int mBase, int nBase,
                                       int kbase, int tid) {
  // A: 128 rows x 32 K of gathered token rows of x (f32 -> bf16)
  const int r4 = tid >> 2;
  const int c0 = (tid & 3) * 8;
#pragma unroll
  for (int h = 0; h < 2; ++h) {
    const int r = r4 + h * 64;
    const int grow = mBase + r;
    uint4 q = make_uint4(0u, 0u, 0u, 0u);
    if (grow < cnt) {
      const int tok = bucket[e * T_TOK + grow] >> 1;
      const float* s = x + (size_t)tok * DDIM + kbase + c0;
      float4 f0 = *(const float4*)(s);
      float4 f1 = *(const float4*)(s + 4);
      q.x = pk2(f0.x, f0.y); q.y = pk2(f0.z, f0.w);
      q.z = pk2(f1.x, f1.y); q.w = pk2(f1.z, f1.w);
    }
    *(uint4*)(sA + r * LSTR + c0) = q;
  }
  // B: 32 K-rows x 64 N-cols of w1/w3, stored transposed [n][k] as bf16
  const int kk = tid >> 3;
  const int n0 = (tid & 7) * 8;
  {
    const float* s = w1p + (size_t)(kbase + kk) * IDIM + nBase + n0;
    __builtin_prefetch(s + 64 * IDIM, 0, 1);  // 2 K-steps ahead (L2 resident)
    float4 f0 = *(const float4*)(s);
    float4 f1 = *(const float4*)(s + 4);
    float v[8] = {f0.x, f0.y, f0.z, f0.w, f1.x, f1.y, f1.z, f1.w};
#pragma unroll
    for (int j = 0; j < 8; ++j) sB1[(n0 + j) * LSTR + kk] = f2bf(v[j]);
  }
  {
    const float* s = w3p + (size_t)(kbase + kk) * IDIM + nBase + n0;
    __builtin_prefetch(s + 64 * IDIM, 0, 1);
    float4 f0 = *(const float4*)(s);
    float4 f1 = *(const float4*)(s + 4);
    float v[8] = {f0.x, f0.y, f0.z, f0.w, f1.x, f1.y, f1.z, f1.w};
#pragma unroll
    for (int j = 0; j < 8; ++j) sB3[(n0 + j) * LSTR + kk] = f2bf(v[j]);
  }
}

__global__ __launch_bounds__(256) void ffn1_kernel(
    const float* __restrict__ x, const float* __restrict__ w1,
    const float* __restrict__ w3, const int* __restrict__ counts,
    const int* __restrict__ offsets, const int* __restrict__ bucket,
    unsigned short* __restrict__ abuf) {
  const int e = blockIdx.z;
  const int cnt = counts[e];
  const int mBase = blockIdx.y * 128;
  if (mBase >= cnt) return;
  const int nBase = blockIdx.x * 64;
  const int aoff = offsets[e];
  const int tid = threadIdx.x;
  const int lane = tid & 31;
  const int wv = tid >> 5;
  const int wm = (wv >> 1) * 32;
  const int wn = (wv & 1) * 32;
  const int lr = lane & 15;
  const int k0 = (lane >> 4) * 8;

  __shared__ __align__(16) unsigned short lsA[2][128 * LSTR];
  __shared__ __align__(16) unsigned short lsB[2][2][64 * LSTR];

  const float* w1p = w1 + (size_t)e * DDIM * IDIM;
  const float* w3p = w3 + (size_t)e * DDIM * IDIM;

  v8f accH[2][2], accG[2][2];
#pragma unroll
  for (int i = 0; i < 2; ++i)
#pragma unroll
    for (int j = 0; j < 2; ++j) {
      accH[i][j] = (v8f){0.f, 0.f, 0.f, 0.f, 0.f, 0.f, 0.f, 0.f};
      accG[i][j] = (v8f){0.f, 0.f, 0.f, 0.f, 0.f, 0.f, 0.f, 0.f};
    }

  stage1(lsA[0], lsB[0][0], lsB[0][1], x, w1p, w3p, bucket, e, cnt, mBase,
         nBase, 0, tid);
  __syncthreads();

  int buf = 0;
  for (int kt = 0; kt < DDIM / 32; ++kt) {
    if (kt + 1 < DDIM / 32)
      stage1(lsA[buf ^ 1], lsB[buf ^ 1][0], lsB[buf ^ 1][1], x, w1p, w3p,
             bucket, e, cnt, mBase, nBase, (kt + 1) * 32, tid);

    Frag a[2], fb1[2], fb3[2];
#pragma unroll
    for (int mi = 0; mi < 2; ++mi) {
      const unsigned short* p = lsA[buf] + (wm + mi * 16 + lr) * LSTR + k0;
      a[mi].q[0] = *(const uint4*)(p);
      a[mi].q[1] = *(const uint4*)(p + 16);
    }
#pragma unroll
    for (int ni = 0; ni < 2; ++ni) {
      const unsigned short* p1 = lsB[buf][0] + (wn + ni * 16 + lr) * LSTR + k0;
      fb1[ni].q[0] = *(const uint4*)(p1);
      fb1[ni].q[1] = *(const uint4*)(p1 + 16);
      const unsigned short* p3 = lsB[buf][1] + (wn + ni * 16 + lr) * LSTR + k0;
      fb3[ni].q[0] = *(const uint4*)(p3);
      fb3[ni].q[1] = *(const uint4*)(p3 + 16);
    }
#pragma unroll
    for (int mi = 0; mi < 2; ++mi)
#pragma unroll
      for (int ni = 0; ni < 2; ++ni) {
        accH[mi][ni] = __builtin_amdgcn_wmma_f32_16x16x32_bf16(
            false, a[mi].v, false, fb1[ni].v, (short)0, accH[mi][ni], false,
            false);
        accG[mi][ni] = __builtin_amdgcn_wmma_f32_16x16x32_bf16(
            false, a[mi].v, false, fb3[ni].v, (short)0, accG[mi][ni], false,
            false);
      }
    __syncthreads();
    buf ^= 1;
  }

  // epilogue: a = silu(h) * g, store bf16 activations compactly
#pragma unroll
  for (int mi = 0; mi < 2; ++mi)
#pragma unroll
    for (int ni = 0; ni < 2; ++ni)
#pragma unroll
      for (int r = 0; r < 8; ++r) {
        const int row = mBase + wm + mi * 16 + (lane >> 4) * 8 + r;
        if (row < cnt) {
          float h = accH[mi][ni][r];
          float g = accG[mi][ni][r];
          float av = (h / (1.f + __expf(-h))) * g;
          const int col = nBase + wn + ni * 16 + lr;
          abuf[(size_t)(aoff + row) * IDIM + col] = f2bf(av);
        }
      }
}

// ---------------------------------------------------------------------------
// Phase 2: Y = A W2 per expert; scatter rows to per-(token,k) buffer.
// A tiles (already bf16) staged global->LDS by the Tensor Data Mover.
// ---------------------------------------------------------------------------
__global__ __launch_bounds__(256) void ffn2_kernel(
    const unsigned short* __restrict__ abuf, const float* __restrict__ w2,
    const int* __restrict__ counts, const int* __restrict__ offsets,
    const int* __restrict__ bucket, float* __restrict__ ybuf) {
  const int e = blockIdx.z;
  const int cnt = counts[e];
  const int mBase = blockIdx.y * 128;
  if (mBase >= cnt) return;
  const int nBase = blockIdx.x * 64;
  const int aoff = offsets[e];
  const int tid = threadIdx.x;
  const int lane = tid & 31;
  const int wv = tid >> 5;
  const int wm = (wv >> 1) * 32;
  const int wn = (wv & 1) * 32;
  const int lr = lane & 15;
  const int k0 = (lane >> 4) * 8;

  __shared__ __align__(16) unsigned short lsA[2][128 * LSTR];
  __shared__ __align__(16) unsigned short lsB[2][64 * LSTR];

  const float* w2p = w2 + (size_t)e * IDIM * DDIM;
  const unsigned short* aRows = abuf + (size_t)(aoff + mBase) * IDIM;
  const u32 rowsAvail = (u32)(cnt - mBase);

  v8f acc[2][2];
#pragma unroll
  for (int i = 0; i < 2; ++i)
#pragma unroll
    for (int j = 0; j < 2; ++j)
      acc[i][j] = (v8f){0.f, 0.f, 0.f, 0.f, 0.f, 0.f, 0.f, 0.f};

  const int r4 = tid >> 2;
  const int c0 = (tid & 3) * 8;
  const int kk = tid >> 3;
  const int n0 = (tid & 7) * 8;
  (void)r4; (void)c0;

  // ---- stage K-step 0 ----
#if USE_TDM
  if (wv == 0)
    tdm_load_tileA((u32)(size_t)(void*)lsA[0], aRows, rowsAvail);
#else
#pragma unroll
  for (int h = 0; h < 2; ++h) {
    const int r = r4 + h * 64;
    uint4 q = make_uint4(0u, 0u, 0u, 0u);
    if (mBase + r < cnt) q = *(const uint4*)(aRows + (size_t)r * IDIM + c0);
    *(uint4*)(lsA[0] + r * LSTR + c0) = q;
  }
#endif
  {
    const float* s = w2p + (size_t)kk * DDIM + nBase + n0;
    __builtin_prefetch(s + 64 * DDIM, 0, 1);
    float4 f0 = *(const float4*)(s);
    float4 f1 = *(const float4*)(s + 4);
    float v[8] = {f0.x, f0.y, f0.z, f0.w, f1.x, f1.y, f1.z, f1.w};
#pragma unroll
    for (int j = 0; j < 8; ++j) lsB[0][(n0 + j) * LSTR + kk] = f2bf(v[j]);
  }
#if USE_TDM
  __builtin_amdgcn_s_wait_tensorcnt(0);
#endif
  __syncthreads();

  int buf = 0;
  for (int kt = 0; kt < IDIM / 32; ++kt) {
    if (kt + 1 < IDIM / 32) {
      const int kbase = (kt + 1) * 32;
#if USE_TDM
      if (wv == 0)
        tdm_load_tileA((u32)(size_t)(void*)lsA[buf ^ 1], aRows + kbase,
                       rowsAvail);
#else
#pragma unroll
      for (int h = 0; h < 2; ++h) {
        const int r = r4 + h * 64;
        uint4 q = make_uint4(0u, 0u, 0u, 0u);
        if (mBase + r < cnt)
          q = *(const uint4*)(aRows + (size_t)r * IDIM + kbase + c0);
        *(uint4*)(lsA[buf ^ 1] + r * LSTR + c0) = q;
      }
#endif
      const float* s = w2p + (size_t)(kbase + kk) * DDIM + nBase + n0;
      __builtin_prefetch(s + 64 * DDIM, 0, 1);
      float4 f0 = *(const float4*)(s);
      float4 f1 = *(const float4*)(s + 4);
      float v[8] = {f0.x, f0.y, f0.z, f0.w, f1.x, f1.y, f1.z, f1.w};
#pragma unroll
      for (int j = 0; j < 8; ++j)
        lsB[buf ^ 1][(n0 + j) * LSTR + kk] = f2bf(v[j]);
    }

    Frag a[2], b[2];
#pragma unroll
    for (int mi = 0; mi < 2; ++mi) {
      const unsigned short* p = lsA[buf] + (wm + mi * 16 + lr) * LSTR + k0;
      a[mi].q[0] = *(const uint4*)(p);
      a[mi].q[1] = *(const uint4*)(p + 16);
    }
#pragma unroll
    for (int ni = 0; ni < 2; ++ni) {
      const unsigned short* p = lsB[buf] + (wn + ni * 16 + lr) * LSTR + k0;
      b[ni].q[0] = *(const uint4*)(p);
      b[ni].q[1] = *(const uint4*)(p + 16);
    }
#pragma unroll
    for (int mi = 0; mi < 2; ++mi)
#pragma unroll
      for (int ni = 0; ni < 2; ++ni)
        acc[mi][ni] = __builtin_amdgcn_wmma_f32_16x16x32_bf16(
            false, a[mi].v, false, b[ni].v, (short)0, acc[mi][ni], false,
            false);
#if USE_TDM
    __builtin_amdgcn_s_wait_tensorcnt(0);  // wave0's TDM copy must land
#endif
    __syncthreads();
    buf ^= 1;
  }

  // scatter rows to per-(token,k) contribution buffer (deterministic)
#pragma unroll
  for (int mi = 0; mi < 2; ++mi)
#pragma unroll
    for (int ni = 0; ni < 2; ++ni)
#pragma unroll
      for (int r = 0; r < 8; ++r) {
        const int row = mBase + wm + mi * 16 + (lane >> 4) * 8 + r;
        if (row < cnt) {
          const int packed = bucket[e * T_TOK + row];  // (t<<1)|k
          const int col = nBase + wn + ni * 16 + lr;
          ybuf[(size_t)packed * DDIM + col] = acc[mi][ni][r];
        }
      }
}

// ---------------------------------------------------------------------------
// Combine: out[t] = w0 * y[t,0] + w1 * y[t,1]
// ---------------------------------------------------------------------------
__global__ void combine_kernel(const float* __restrict__ ybuf,
                               const float* __restrict__ wbuf,
                               float* __restrict__ out) {
  const int t = blockIdx.x;
  const float w0 = wbuf[t * 2 + 0];
  const float w1 = wbuf[t * 2 + 1];
  const float* y0 = ybuf + (size_t)(t * 2) * DDIM;
  const float* y1 = y0 + DDIM;
  float* o = out + (size_t)t * DDIM;
  for (int d = threadIdx.x; d < DDIM; d += blockDim.x)
    o[d] = w0 * y0[d] + w1 * y1[d];
}

// ---------------------------------------------------------------------------
extern "C" void kernel_launch(void* const* d_in, const int* in_sizes, int n_in,
                              void* d_out, int out_size, void* d_ws,
                              size_t ws_size, hipStream_t stream) {
  (void)in_sizes; (void)n_in; (void)out_size; (void)ws_size;
  const float* x  = (const float*)d_in[0];  // [2,2048,1024]
  const float* rw = (const float*)d_in[1];  // [8,1024]
  const float* w1 = (const float*)d_in[2];  // [8,1024,1024]
  const float* w2 = (const float*)d_in[3];  // [8,1024,1024]
  const float* w3 = (const float*)d_in[4];  // [8,1024,1024]

  float* out    = (float*)d_out;                      // [T*D] out
  float* logits = out + (size_t)T_TOK * DDIM;         // then [T*E] logits

  char* ws = (char*)d_ws;
  int*   counts  = (int*)(ws + 0);
  int*   offsets = (int*)(ws + 64);
  float* wbuf    = (float*)(ws + 128);                // T*2 f32 (32KB)
  int*   bucket  = (int*)(ws + 128 + 32768);          // E*T i32 (128KB)
  unsigned short* abuf = (unsigned short*)(ws + 262144);            // 2T*I bf16 (16MB)
  float* ybuf = (float*)(ws + 262144 + (size_t)2 * T_TOK * IDIM * 2); // 2T*D f32 (32MB)

  init_kernel<<<1, 32, 0, stream>>>(counts);
  router_kernel<<<T_TOK / 8, 256, 0, stream>>>(x, rw, logits, wbuf, bucket,
                                               counts);
  scan_kernel<<<1, 32, 0, stream>>>(counts, offsets);

  dim3 g1(IDIM / 64, T_TOK / 128, NEXP);
  ffn1_kernel<<<g1, 256, 0, stream>>>(x, w1, w3, counts, offsets, bucket, abuf);

  dim3 g2(DDIM / 64, T_TOK / 128, NEXP);
  ffn2_kernel<<<g2, 256, 0, stream>>>(abuf, w2, counts, offsets, bucket, ybuf);

  combine_kernel<<<T_TOK, 256, 0, stream>>>(ybuf, wbuf, out);
}
